// DimeNetPlusPlusWrap_13142599926312
// MI455X (gfx1250) — matrix operand order repaired and checked by reading, not compiled, not run on hardware
//
#include <hip/hip_runtime.h>
#include <hip/hip_bf16.h>

// ---------------- problem constants (fixed by the reference) ----------------
#define E_EDGES   200000
#define T_TRIP    2000000
#define N_NODES   20000
#define HDIM      128
#define INTC      64
#define BASC      8
#define OUTE      256
#define NRADC     6
#define SBFD      42
#define NBLK      4

// ---------------- bf16 helpers (storage-only, manual RNE round) -------------
typedef __bf16 bf16_t;
typedef __attribute__((ext_vector_type(16))) __bf16 v16bf;
typedef __attribute__((ext_vector_type(8)))  __bf16 v8bf;
typedef __attribute__((ext_vector_type(8)))  float  v8f;

__device__ __forceinline__ unsigned int f2bf_bits(float f) {
    unsigned int u = __builtin_bit_cast(unsigned int, f);
    unsigned int r = u + 0x7FFFu + ((u >> 16) & 1u);   // round-to-nearest-even
    return r >> 16;
}
__device__ __forceinline__ unsigned int pack2(float lo, float hi) {
    return f2bf_bits(lo) | (f2bf_bits(hi) << 16);
}
__device__ __forceinline__ float bf2f(unsigned short s) {
    return __builtin_bit_cast(float, (unsigned int)s << 16);
}
__device__ __forceinline__ unsigned int mulpk(unsigned int a, unsigned int m) {
    float a0 = bf2f((unsigned short)a), a1 = bf2f((unsigned short)(a >> 16));
    float m0 = bf2f((unsigned short)m), m1 = bf2f((unsigned short)(m >> 16));
    return pack2(a0 * m0, a1 * m1);
}
__device__ __forceinline__ uint4 mulpk4(uint4 a, uint4 m) {
    uint4 r; r.x = mulpk(a.x, m.x); r.y = mulpk(a.y, m.y);
    r.z = mulpk(a.z, m.z); r.w = mulpk(a.w, m.w); return r;
}
__device__ __forceinline__ float silu(float v) {
    return v / (1.0f + __expf(-v));
}
// Build a v16bf fragment from two 16-byte-aligned LDS runs (2x ds_load_b128).
__device__ __forceinline__ v16bf frag2(const unsigned short* p0,
                                       const unsigned short* p1) {
    v8bf lo = *reinterpret_cast<const v8bf*>(p0);
    v8bf hi = *reinterpret_cast<const v8bf*>(p1);
    return __builtin_shufflevector(lo, hi, 0,1,2,3,4,5,6,7,8,9,10,11,12,13,14,15);
}
// CDNA5 async global->LDS 128-bit copy (ISA 10., opcode 98; ASYNCcnt-tracked).
__device__ __forceinline__ void async_copy_b128(unsigned lds_off, const void* gptr) {
    asm volatile("global_load_async_to_lds_b128 %0, %1, off"
                 :: "v"(lds_off), "v"(gptr) : "memory");
}
__device__ __forceinline__ void wait_async() {
    asm volatile("s_wait_asynccnt 0x0" ::: "memory");
}

// fp32 -> bf16 bulk converter (weights once per launch, x once)
__global__ __launch_bounds__(256)
void cvt_f32_bf16(const float* __restrict__ s, unsigned short* __restrict__ d,
                  long long n)
{
    const long long i = (long long)blockIdx.x * 256 + threadIdx.x;
    if (i < n) d[i] = (unsigned short)f2bf_bits(s[i]);
}

// ---------------- generic WMMA GEMM: C = epi(A[M,K] @ B[K,N]) ---------------
// A: bf16 (AF32=0, K multiple of 8) or fp32 (AF32=1).  B: bf16 (preconverted).
// C/Res: bf16.  bias: fp32.
// Tile: 128(M) x 64(N), 8 waves (4 row x 2 col), each wave 32x32 = 4 WMMAs
// per K-step of 32.  LDS rows 80B stride (16B-aligned, conflict-free):
//   As[row][k] row-major    -> A-frag = 2 contiguous 8-elem runs
//   Bt[col][k] K-contiguous -> B-frag = 2 contiguous 8-elem runs
#define GT_M 128
#define GT_N 64
#define GT_K 32
#define LKA  40   // padded k-stride in shorts (80 bytes)

template<int ACT, int BIAS, int RES, int AMUL, int AF32>
__global__ __launch_bounds__(256)
void gemm_wmma(const void* __restrict__ Ap, const void* __restrict__ Amulp,
               const unsigned short* __restrict__ B, const float* __restrict__ bias,
               const unsigned short* __restrict__ Res, unsigned short* __restrict__ C,
               int M, int N, int K)
{
    __shared__ __align__(16) unsigned short As[GT_M][LKA];
    __shared__ __align__(16) unsigned short Bt[GT_N][LKA];

    const int tid  = threadIdx.x;
    const int lane = tid & 31;
    const int wave = tid >> 5;
    const int wm   = wave & 3;
    const int wn   = wave >> 2;
    const int half = (lane >> 4) & 1;
    const int rr   = lane & 15;

    const long long rowBase = (long long)blockIdx.x * GT_M;
    const long long colBase = (long long)blockIdx.y * GT_N;

    // Async (TDM-path) staging is safe only when K%32==0: otherwise stale LDS
    // in the K-pad could inject NaN*0 into valid accumulators.
    const bool kAligned = (K % GT_K) == 0;

    v8f a00 = {}, a01 = {}, a10 = {}, a11 = {};

    for (int k0 = 0; k0 < K; k0 += GT_K) {
        // ---- A tile: thread -> row = tid/2, 16 contiguous k
        {
            const int r  = tid >> 1;
            const int c0 = (tid & 1) * 16;
            const long long gr = rowBase + r;
            if constexpr (!AF32 && !AMUL) {
                const unsigned short* Ab = (const unsigned short*)Ap;
                if (kAligned) {
                    // pure byte copy: global bf16 -> LDS via async-to-LDS.
                    // Rows >= M simply don't issue; their stale LDS only feeds
                    // output rows the epilogue never stores.
                    if (gr < M) {
                        const long long base = gr * (long long)K + k0 + c0;
                        async_copy_b128((unsigned)(unsigned long long)&As[r][c0],
                                        (const void*)&Ab[base]);
                        async_copy_b128((unsigned)(unsigned long long)&As[r][c0 + 8],
                                        (const void*)&Ab[base + 8]);
                    }
                } else {
                    uint4 u0 = {0,0,0,0}, u1 = {0,0,0,0};
                    if (gr < M) {
                        const long long base = gr * (long long)K + k0 + c0;
                        if (k0 + c0 + 8  <= K) u0 = *reinterpret_cast<const uint4*>(&Ab[base]);
                        if (k0 + c0 + 16 <= K) u1 = *reinterpret_cast<const uint4*>(&Ab[base + 8]);
                    }
                    *reinterpret_cast<uint4*>(&As[r][c0])     = u0;
                    *reinterpret_cast<uint4*>(&As[r][c0 + 8]) = u1;
                }
            } else if constexpr (!AF32) {
                // bf16 A with fused elementwise multiplier (register path)
                const unsigned short* Ab = (const unsigned short*)Ap;
                const unsigned short* Mb = (const unsigned short*)Amulp;
                uint4 u0 = {0,0,0,0}, u1 = {0,0,0,0};
                uint4 m0 = {0,0,0,0}, m1 = {0,0,0,0};
                if (gr < M) {
                    const long long base = gr * (long long)K + k0 + c0;
                    if (k0 + c0 + 8 <= K) {
                        u0 = *reinterpret_cast<const uint4*>(&Ab[base]);
                        m0 = *reinterpret_cast<const uint4*>(&Mb[base]);
                    }
                    if (k0 + c0 + 16 <= K) {
                        u1 = *reinterpret_cast<const uint4*>(&Ab[base + 8]);
                        m1 = *reinterpret_cast<const uint4*>(&Mb[base + 8]);
                    }
                }
                u0 = mulpk4(u0, m0);
                u1 = mulpk4(u1, m1);
                *reinterpret_cast<uint4*>(&As[r][c0])     = u0;
                *reinterpret_cast<uint4*>(&As[r][c0 + 8]) = u1;
            } else {
                // fp32 A (atomics buffers / raw inputs), convert + pack
                const float* Af = (const float*)Ap;
                unsigned int pk[8];
#pragma unroll
                for (int j = 0; j < 8; ++j) {
                    const int kk = k0 + c0 + 2 * j;
                    float v0 = 0.0f, v1 = 0.0f;
                    if (gr < M) {
                        const long long base = gr * (long long)K;
                        if (kk < K)     v0 = Af[base + kk];
                        if (kk + 1 < K) v1 = Af[base + kk + 1];
                    }
                    pk[j] = pack2(v0, v1);
                }
                uint4 u0; u0.x = pk[0]; u0.y = pk[1]; u0.z = pk[2]; u0.w = pk[3];
                uint4 u1; u1.x = pk[4]; u1.y = pk[5]; u1.z = pk[6]; u1.w = pk[7];
                *reinterpret_cast<uint4*>(&As[r][c0])     = u0;
                *reinterpret_cast<uint4*>(&As[r][c0 + 8]) = u1;
            }
        }
        // ---- B tile (bf16 weights): coalesced uint4 load + transposed scatter
        {
            const int kr = tid >> 3;          // 0..31
            const int c0 = (tid & 7) * 8;     // 0..56
            const int kk = k0 + kr;
            uint4 u = {0,0,0,0};
            const long long gc0 = colBase + c0;
            if (kk < K && gc0 + 8 <= N)
                u = *reinterpret_cast<const uint4*>(&B[(long long)kk * N + gc0]);
            Bt[c0 + 0][kr] = (unsigned short)(u.x);
            Bt[c0 + 1][kr] = (unsigned short)(u.x >> 16);
            Bt[c0 + 2][kr] = (unsigned short)(u.y);
            Bt[c0 + 3][kr] = (unsigned short)(u.y >> 16);
            Bt[c0 + 4][kr] = (unsigned short)(u.z);
            Bt[c0 + 5][kr] = (unsigned short)(u.z >> 16);
            Bt[c0 + 6][kr] = (unsigned short)(u.w);
            Bt[c0 + 7][kr] = (unsigned short)(u.w >> 16);
        }
        if constexpr (!AF32 && !AMUL) wait_async();
        __syncthreads();

        const int h8  = half * 8;
        const int h16 = half * 16;
        v16bf am0 = frag2(&As[wm * 32 + rr][h8],      &As[wm * 32 + rr][16 + h8]);
        v16bf am1 = frag2(&As[wm * 32 + 16 + rr][h8], &As[wm * 32 + 16 + rr][16 + h8]);
        v16bf bn0 = frag2(&Bt[wn * 32 + rr][h8],      &Bt[wn * 32 + rr][16 + h8]);
        v16bf bn1 = frag2(&Bt[wn * 32 + 16 + rr][h8], &Bt[wn * 32 + 16 + rr][16 + h8]);

        a00 = __builtin_amdgcn_wmma_f32_16x16x32_bf16(false, am0, false, bn0, (short)0, a00, false, false);
        a01 = __builtin_amdgcn_wmma_f32_16x16x32_bf16(false, am0, false, bn1, (short)0, a01, false, false);
        a10 = __builtin_amdgcn_wmma_f32_16x16x32_bf16(false, am1, false, bn0, (short)0, a10, false, false);
        a11 = __builtin_amdgcn_wmma_f32_16x16x32_bf16(false, am1, false, bn1, (short)0, a11, false, false);
        __syncthreads();
    }

    // epilogue: row = base + mi*16 + vg + half*8, col = base + ni*16 + rr
#pragma unroll
    for (int mi = 0; mi < 2; ++mi) {
#pragma unroll
        for (int vg = 0; vg < 8; ++vg) {
            const long long r = rowBase + wm * 32 + mi * 16 + vg + half * 8;
            if (r >= M) continue;
#pragma unroll
            for (int ni = 0; ni < 2; ++ni) {
                const long long c = colBase + wn * 32 + ni * 16 + rr;
                if (c >= N) continue;
                float v = mi == 0 ? (ni == 0 ? a00[vg] : a01[vg])
                                  : (ni == 0 ? a10[vg] : a11[vg]);
                if (BIAS) v += bias[c];
                if (ACT)  v = silu(v);
                const long long off = r * (long long)N + c;
                if (RES)  v += bf2f(Res[off]);
                C[off] = (unsigned short)f2bf_bits(v);
            }
        }
    }
}

static void launch_gemm(hipStream_t s,
                        const void* A, const void* Amul, const unsigned short* B,
                        const float* bias, const unsigned short* Res, unsigned short* C,
                        int M, int N, int K,
                        int act, int bia, int res, int amul, int af32)
{
    dim3 g((M + GT_M - 1) / GT_M, (N + GT_N - 1) / GT_N), b(256);
    if (af32) {
        if (act && !bia && res && !amul)
            gemm_wmma<1,0,1,0,1><<<g, b, 0, s>>>(A, Amul, B, bias, Res, C, M, N, K);
        else if (!act && bia && !res && !amul)
            gemm_wmma<0,1,0,0,1><<<g, b, 0, s>>>(A, Amul, B, bias, Res, C, M, N, K);
        else
            gemm_wmma<0,0,0,0,1><<<g, b, 0, s>>>(A, Amul, B, bias, Res, C, M, N, K);
    } else {
        if (act && bia && !res && !amul)
            gemm_wmma<1,1,0,0,0><<<g, b, 0, s>>>(A, Amul, B, bias, Res, C, M, N, K);
        else if (act && !bia && !res && amul)
            gemm_wmma<1,0,0,1,0><<<g, b, 0, s>>>(A, Amul, B, bias, Res, C, M, N, K);
        else if (act && bia && res && !amul)
            gemm_wmma<1,1,1,0,0><<<g, b, 0, s>>>(A, Amul, B, bias, Res, C, M, N, K);
        else
            gemm_wmma<0,0,0,0,0><<<g, b, 0, s>>>(A, Amul, B, bias, Res, C, M, N, K);
    }
}

// ---------------- fused triplet kernel -------------------------------------
// Per 128-triplet tile: sbf_t = (sbf @ Wsbf1) @ Wsbf2 via two WMMA stages
// chained through LDS (K padded 42->64 and 8->32), then
// m = sbf_t * gather(x_kj_down[bf16], idx_kj) scattered into agg[idx_ji] with
// fp32 atomics.  x_kj_down (26MB bf16) and agg (51MB) stay in the 192MB L2.
#define TTILE 128
#define LKS   72   // Ss/W1t k-stride in shorts (144B, 16B-aligned)

__global__ __launch_bounds__(256)
void triplet_kernel(const float*          __restrict__ sbf,    // [T,42] fp32
                    const unsigned short* __restrict__ Wsbf1,  // [42,8] bf16
                    const unsigned short* __restrict__ Wsbf2,  // [8,64] bf16
                    const unsigned short* __restrict__ xkj,    // [E,64] bf16
                    const int*            __restrict__ idx_kj,
                    const int*            __restrict__ idx_ji,
                    float*                __restrict__ agg,    // [E,64] fp32
                    int Ttot)
{
    __shared__ __align__(16) unsigned short Ss [TTILE][LKS]; // sbf tile, K pad 64
    __shared__ __align__(16) unsigned short W1t[16][LKS];    // Wsbf1^T, K pad 64
    __shared__ __align__(16) unsigned short W2t[64][LKA];    // Wsbf2^T, K pad 32
    __shared__ __align__(16) unsigned short T1s[TTILE][LKA]; // stage-1, K pad 32

    const int tid  = threadIdx.x;
    const int lane = tid & 31;
    const int wave = tid >> 5;
    const int half = (lane >> 4) & 1;
    const int rr   = lane & 15;
    const int h8   = half * 8;
    const int h16  = half * 16;
    const long long tBase = (long long)blockIdx.x * TTILE;

    // sbf tile: thread -> row = tid/2, 32 contiguous (padded) k, 4x b128 store
    {
        const int r  = tid >> 1;
        const int c0 = (tid & 1) * 32;
        const long long g = tBase + r;
#pragma unroll
        for (int q = 0; q < 4; ++q) {
            unsigned int pk[4];
#pragma unroll
            for (int j = 0; j < 4; ++j) {
                const int c = c0 + q * 8 + 2 * j;
                float v0 = (g < Ttot && c     < SBFD) ? sbf[g * SBFD + c]     : 0.0f;
                float v1 = (g < Ttot && c + 1 < SBFD) ? sbf[g * SBFD + c + 1] : 0.0f;
                pk[j] = pack2(v0, v1);
            }
            uint4 u; u.x = pk[0]; u.y = pk[1]; u.z = pk[2]; u.w = pk[3];
            *reinterpret_cast<uint4*>(&Ss[r][c0 + q * 8]) = u;
        }
    }
    // W1^T fill (bf16 source, race-free single pass incl. zero pad)
    for (int i = tid; i < 16 * 64; i += 256) {
        const int c = i >> 6, k = i & 63;
        W1t[c][k] = (k < SBFD && c < BASC) ? Wsbf1[k * BASC + c] : (unsigned short)0;
    }
    // W2^T fill
    for (int i = tid; i < 64 * 32; i += 256) {
        const int c = i >> 5, k = i & 31;
        W2t[c][k] = (k < BASC) ? Wsbf2[k * INTC + c] : (unsigned short)0;
    }
    // zero K-pad region of T1s (cols 16..31); stage-1 writes cols 0..15
    for (int i = tid; i < TTILE * 16; i += 256) {
        T1s[i >> 4][16 + (i & 15)] = 0;
    }
    __syncthreads();

    // stage 1: tmp[128,16] = Ss[128,64] @ Wsbf1[64,16]
    {
        v8f acc = {};
#pragma unroll
        for (int k0 = 0; k0 < 64; k0 += 32) {
            v16bf af = frag2(&Ss[wave * 16 + rr][k0 + h8], &Ss[wave * 16 + rr][k0 + 16 + h8]);
            v16bf bv = frag2(&W1t[rr][k0 + h8],           &W1t[rr][k0 + 16 + h8]);
            acc = __builtin_amdgcn_wmma_f32_16x16x32_bf16(false, af, false, bv,
                                                          (short)0, acc, false, false);
        }
#pragma unroll
        for (int vg = 0; vg < 8; ++vg)
            T1s[wave * 16 + vg + half * 8][rr] = (unsigned short)f2bf_bits(acc[vg]);
    }
    __syncthreads();

    // stage 2: sbf_t[128,64] = tmp[128,8(pad32)] @ Wsbf2[32,64]
    v8f acc2[4] = {v8f{}, v8f{}, v8f{}, v8f{}};
    v16bf af2 = frag2(&T1s[wave * 16 + rr][h8], &T1s[wave * 16 + rr][16 + h8]);
#pragma unroll
    for (int ns = 0; ns < 4; ++ns) {
        v16bf bv = frag2(&W2t[ns * 16 + rr][h8], &W2t[ns * 16 + rr][16 + h8]);
        acc2[ns] = __builtin_amdgcn_wmma_f32_16x16x32_bf16(false, af2, false, bv,
                                                           (short)0, acc2[ns], false, false);
    }

    // gather bf16 x_kj rows, multiply, scatter-add into agg (L2 atomics)
#pragma unroll
    for (int vg = 0; vg < 8; ++vg) {
        const long long t = tBase + wave * 16 + vg + half * 8;
        if (t < Ttot) {
            const long long ekj = idx_kj[t];
            const long long eji = idx_ji[t];
#pragma unroll
            for (int ns = 0; ns < 4; ++ns) {
                const int c = ns * 16 + rr;
                const float m = acc2[ns][vg] * bf2f(xkj[ekj * INTC + c]);
                atomicAdd(&agg[eji * INTC + c], m);
            }
        }
    }
}

// ---------------- output block helpers -------------------------------------
// nacc[idx_i[e]][h] += (rbf[e] . Wo_rbf[:,h]) * x[e][h]   (K=6, VALU + atomics)
__global__ __launch_bounds__(256)
void out_rbf_scatter(const float*          __restrict__ rbf,   // [E,6] fp32
                     const float*          __restrict__ Wor,   // [6,128] fp32
                     const unsigned short* __restrict__ x,     // [E,128] bf16
                     const int*            __restrict__ idx_i,
                     float* __restrict__ nacc)                 // [N,128] fp32
{
    __shared__ float Ws[NRADC * HDIM];
    const int tid = threadIdx.x;
    for (int i = tid; i < NRADC * HDIM; i += 256) Ws[i] = Wor[i];
    __syncthreads();
    const long long e = (long long)blockIdx.x * 2 + (tid >> 7);
    const int h = tid & 127;
    if (e < E_EDGES) {
        float s = 0.0f;
#pragma unroll
        for (int k = 0; k < NRADC; ++k) s += rbf[e * NRADC + k] * Ws[k * HDIM + h];
        const float v = s * bf2f(x[e * HDIM + h]);
        atomicAdd(&nacc[(long long)idx_i[e] * HDIM + h], v);
    }
}

// out[n] += t[n,:] . Wo_out   (wave-per-node shuffle reduction)
__global__ __launch_bounds__(256)
void out_proj_add(const unsigned short* __restrict__ t,  // [N,256] bf16
                  const float* __restrict__ Wo,          // [256] fp32
                  float* __restrict__ out, int Nn)
{
    const int gid  = blockIdx.x * blockDim.x + threadIdx.x;
    const int n    = gid >> 5;
    const int lane = gid & 31;
    if (n >= Nn) return;
    float s = 0.0f;
    for (int c = lane; c < OUTE; c += 32) s += bf2f(t[(long long)n * OUTE + c]) * Wo[c];
    for (int o = 16; o > 0; o >>= 1) s += __shfl_down(s, o, 32);
    if (lane == 0) out[n] += s;
}

// ---------------- host orchestration ---------------------------------------
static void cvt(hipStream_t s, const float* src, unsigned short* dst, long long n) {
    cvt_f32_bf16<<<(unsigned)((n + 255) / 256), 256, 0, s>>>(src, dst, n);
}

static void output_block(hipStream_t s, int ob, const unsigned short* x,
                         const float* rbf, const int* idx_i, const float* Wo_rbf,
                         const unsigned short* Woupb, const float* bo_up,
                         const unsigned short* Wolinsb, const float* bo_lins,
                         const float* Wo_out, float* nacc,
                         unsigned short* nt1, unsigned short* nt2, float* out)
{
    hipMemsetAsync(nacc, 0, (size_t)N_NODES * HDIM * sizeof(float), s);
    out_rbf_scatter<<<E_EDGES / 2, 256, 0, s>>>(rbf, Wo_rbf + (size_t)ob * NRADC * HDIM,
                                                x, idx_i, nacc);
    launch_gemm(s, nacc, nullptr, Woupb + (size_t)ob * HDIM * OUTE,
                bo_up + (size_t)ob * OUTE, nullptr, nt1,
                N_NODES, OUTE, HDIM, 0, 1, 0, 0, 1);
    unsigned short* a = nt1; unsigned short* c = nt2;
    for (int l = 0; l < 3; ++l) {
        launch_gemm(s, a, nullptr, Wolinsb + (size_t)(ob * 3 + l) * OUTE * OUTE,
                    bo_lins + (size_t)(ob * 3 + l) * OUTE, nullptr, c,
                    N_NODES, OUTE, OUTE, 1, 1, 0, 0, 0);
        unsigned short* tmp = a; a = c; c = tmp;
    }
    out_proj_add<<<(N_NODES * 32 + 255) / 256, 256, 0, s>>>(
        a, Wo_out + (size_t)ob * OUTE, out, N_NODES);
}

extern "C" void kernel_launch(void* const* d_in, const int* in_sizes, int n_in,
                              void* d_out, int out_size, void* d_ws, size_t ws_size,
                              hipStream_t stream)
{
    (void)in_sizes; (void)n_in; (void)out_size; (void)ws_size;
    const float* x0     = (const float*)d_in[0];
    const float* rbf    = (const float*)d_in[1];
    const float* sbf    = (const float*)d_in[2];
    const float* W_rbf1 = (const float*)d_in[3];
    const float* W_rbf2 = (const float*)d_in[4];
    const float* W_sbf1 = (const float*)d_in[5];
    const float* W_sbf2 = (const float*)d_in[6];
    const float* W_kj   = (const float*)d_in[7];
    const float* b_kj   = (const float*)d_in[8];
    const float* W_ji   = (const float*)d_in[9];
    const float* b_ji   = (const float*)d_in[10];
    const float* W_down = (const float*)d_in[11];
    const float* W_up   = (const float*)d_in[12];
    const float* Wb     = (const float*)d_in[13];
    const float* bb     = (const float*)d_in[14];
    const float* Wa     = (const float*)d_in[15];
    const float* ba     = (const float*)d_in[16];
    const float* W_lin  = (const float*)d_in[17];
    const float* b_lin  = (const float*)d_in[18];
    const float* Wo_rbf = (const float*)d_in[19];
    const float* Wo_up  = (const float*)d_in[20];
    const float* bo_up  = (const float*)d_in[21];
    const float* Wo_lins= (const float*)d_in[22];
    const float* bo_lins= (const float*)d_in[23];
    const float* Wo_out = (const float*)d_in[24];
    const int*   idx_kj = (const int*)d_in[25];
    const int*   idx_ji = (const int*)d_in[26];
    const int*   idx_i  = (const int*)d_in[27];
    float* out = (float*)d_out;

    // ---- workspace carving ----
    char* ws = (char*)d_ws;
    size_t off = 0;
    auto carveB = [&](size_t bytes) -> char* {
        char* p = ws + off;
        off = (off + bytes + 255) & ~(size_t)255;
        return p;
    };
    const size_t EHb = (size_t)E_EDGES * HDIM * sizeof(unsigned short);
    unsigned short* xin  = (unsigned short*)carveB(EHb);
    unsigned short* bufJ = (unsigned short*)carveB(EHb);
    unsigned short* bufK = (unsigned short*)carveB(EHb);
    unsigned short* bufH = (unsigned short*)carveB(EHb);
    unsigned short* xA   = (unsigned short*)carveB(EHb);
    unsigned short* xB   = (unsigned short*)carveB(EHb);
    unsigned short* buf8 = (unsigned short*)carveB((size_t)E_EDGES * BASC * 2);
    unsigned short* bufD = (unsigned short*)carveB((size_t)E_EDGES * INTC * 2);
    float* aggb = (float*)carveB((size_t)E_EDGES * INTC * sizeof(float));
    float* nacc = (float*)carveB((size_t)N_NODES * HDIM * sizeof(float));
    unsigned short* nt1 = (unsigned short*)carveB((size_t)N_NODES * OUTE * 2);
    unsigned short* nt2 = (unsigned short*)carveB((size_t)N_NODES * OUTE * 2);
    // bf16 weight copies
    unsigned short* Wrbf1b = (unsigned short*)carveB((size_t)NBLK * NRADC * BASC * 2);
    unsigned short* Wrbf2b = (unsigned short*)carveB((size_t)NBLK * BASC * HDIM * 2);
    unsigned short* Wsbf1b = (unsigned short*)carveB((size_t)NBLK * SBFD * BASC * 2);
    unsigned short* Wsbf2b = (unsigned short*)carveB((size_t)NBLK * BASC * INTC * 2);
    unsigned short* Wkjb   = (unsigned short*)carveB((size_t)NBLK * HDIM * HDIM * 2);
    unsigned short* Wjib   = (unsigned short*)carveB((size_t)NBLK * HDIM * HDIM * 2);
    unsigned short* Wdownb = (unsigned short*)carveB((size_t)NBLK * HDIM * INTC * 2);
    unsigned short* Wupb   = (unsigned short*)carveB((size_t)NBLK * INTC * HDIM * 2);
    unsigned short* Wbb    = (unsigned short*)carveB((size_t)NBLK * 2 * HDIM * HDIM * 2);
    unsigned short* Wab    = (unsigned short*)carveB((size_t)NBLK * 4 * HDIM * HDIM * 2);
    unsigned short* Wlinb  = (unsigned short*)carveB((size_t)NBLK * HDIM * HDIM * 2);
    unsigned short* Woupb  = (unsigned short*)carveB((size_t)(NBLK + 1) * HDIM * OUTE * 2);
    unsigned short* Wolinsb= (unsigned short*)carveB((size_t)(NBLK + 1) * 3 * OUTE * OUTE * 2);

    // ---- one-time conversions (replayed per launch; deterministic) ----
    cvt(stream, x0,     xin,    (long long)E_EDGES * HDIM);
    cvt(stream, W_rbf1, Wrbf1b, (long long)NBLK * NRADC * BASC);
    cvt(stream, W_rbf2, Wrbf2b, (long long)NBLK * BASC * HDIM);
    cvt(stream, W_sbf1, Wsbf1b, (long long)NBLK * SBFD * BASC);
    cvt(stream, W_sbf2, Wsbf2b, (long long)NBLK * BASC * INTC);
    cvt(stream, W_kj,   Wkjb,   (long long)NBLK * HDIM * HDIM);
    cvt(stream, W_ji,   Wjib,   (long long)NBLK * HDIM * HDIM);
    cvt(stream, W_down, Wdownb, (long long)NBLK * HDIM * INTC);
    cvt(stream, W_up,   Wupb,   (long long)NBLK * INTC * HDIM);
    cvt(stream, Wb,     Wbb,    (long long)NBLK * 2 * HDIM * HDIM);
    cvt(stream, Wa,     Wab,    (long long)NBLK * 4 * HDIM * HDIM);
    cvt(stream, W_lin,  Wlinb,  (long long)NBLK * HDIM * HDIM);
    cvt(stream, Wo_up,  Woupb,  (long long)(NBLK + 1) * HDIM * OUTE);
    cvt(stream, Wo_lins,Wolinsb,(long long)(NBLK + 1) * 3 * OUTE * OUTE);

    hipMemsetAsync(out, 0, (size_t)N_NODES * sizeof(float), stream);

    // output block 0 on the original x
    output_block(stream, 0, xin, rbf, idx_i, Wo_rbf, Woupb, bo_up,
                 Wolinsb, bo_lins, Wo_out, nacc, nt1, nt2, out);

    const unsigned short* xc = xin;
    unsigned short* ping[2] = { xA, xB };
    for (int b = 0; b < NBLK; ++b) {
        unsigned short* xn = ping[b & 1];
        // x_ji = silu(x @ W_ji + b_ji) ; x_kj = silu(x @ W_kj + b_kj)
        launch_gemm(stream, xc, nullptr, Wjib + (size_t)b * HDIM * HDIM,
                    b_ji + (size_t)b * HDIM, nullptr, bufJ, E_EDGES, HDIM, HDIM, 1,1,0,0,0);
        launch_gemm(stream, xc, nullptr, Wkjb + (size_t)b * HDIM * HDIM,
                    b_kj + (size_t)b * HDIM, nullptr, bufK, E_EDGES, HDIM, HDIM, 1,1,0,0,0);
        // rbf_t = (rbf @ W_rbf1) @ W_rbf2   (rbf is fp32, K=6)
        launch_gemm(stream, rbf, nullptr, Wrbf1b + (size_t)b * NRADC * BASC,
                    nullptr, nullptr, buf8, E_EDGES, BASC, NRADC, 0,0,0,0,1);
        launch_gemm(stream, buf8, nullptr, Wrbf2b + (size_t)b * BASC * HDIM,
                    nullptr, nullptr, bufH, E_EDGES, HDIM, BASC, 0,0,0,0,0);
        // x_kj_down = silu((x_kj * rbf_t) @ W_down)   [AMUL fusion]
        launch_gemm(stream, bufK, bufH, Wdownb + (size_t)b * HDIM * INTC,
                    nullptr, nullptr, bufD, E_EDGES, INTC, HDIM, 1,0,0,1,0);
        // triplet: agg = segsum(x_kj_down[idx_kj] * sbf_t, idx_ji)
        hipMemsetAsync(aggb, 0, (size_t)E_EDGES * INTC * sizeof(float), stream);
        triplet_kernel<<<T_TRIP / TTILE, 256, 0, stream>>>(
            sbf, Wsbf1b + (size_t)b * SBFD * BASC, Wsbf2b + (size_t)b * BASC * INTC,
            bufD, idx_kj, idx_ji, aggb, T_TRIP);
        // h = x_ji + silu(agg @ W_up)   (agg fp32)
        launch_gemm(stream, aggb, nullptr, Wupb + (size_t)b * INTC * HDIM,
                    nullptr, bufJ, bufH, E_EDGES, HDIM, INTC, 1,0,1,0,1);
        // residual layer before skip
        launch_gemm(stream, bufH, nullptr, Wbb + (size_t)(b * 2 + 0) * HDIM * HDIM,
                    bb + (size_t)(b * 2 + 0) * HDIM, nullptr, bufJ, E_EDGES, HDIM, HDIM, 1,1,0,0,0);
        launch_gemm(stream, bufJ, nullptr, Wbb + (size_t)(b * 2 + 1) * HDIM * HDIM,
                    bb + (size_t)(b * 2 + 1) * HDIM, bufH, bufH, E_EDGES, HDIM, HDIM, 1,1,1,0,0);
        // skip: x_next = silu(h @ W_lin + b_lin) + x
        launch_gemm(stream, bufH, nullptr, Wlinb + (size_t)b * HDIM * HDIM,
                    b_lin + (size_t)b * HDIM, xc, xn, E_EDGES, HDIM, HDIM, 1,1,1,0,0);
        // two residual layers after skip (in place on x_next)
        for (int l = 0; l < 2; ++l) {
            launch_gemm(stream, xn, nullptr, Wab + (size_t)(b * 4 + 2 * l) * HDIM * HDIM,
                        ba + (size_t)(b * 4 + 2 * l) * HDIM, nullptr, bufJ,
                        E_EDGES, HDIM, HDIM, 1,1,0,0,0);
            launch_gemm(stream, bufJ, nullptr, Wab + (size_t)(b * 4 + 2 * l + 1) * HDIM * HDIM,
                        ba + (size_t)(b * 4 + 2 * l + 1) * HDIM, xn, xn,
                        E_EDGES, HDIM, HDIM, 1,1,1,0,0);
        }
        xc = xn;
        output_block(stream, b + 1, xc, rbf, idx_i, Wo_rbf, Woupb, bo_up,
                     Wolinsb, bo_lins, Wo_out, nacc, nt1, nt2, out);
    }
}
